// AttnProcessor_5617817223506
// MI455X (gfx1250) — compile-verified, matrix operand order checked by reading, not run on hardware
//
#include <hip/hip_runtime.h>
#include <hip/hip_bf16.h>
#include <cstdint>

// ---------- fixed problem geometry ----------
#define BATCH   2
#define SQ      2048
#define SKV     2048
#define CDIM    1280
#define CCDIM   1024
#define HEADS   20
#define DHEAD   64
#define MROWS   (BATCH * SQ)       // 4096

typedef __bf16 bf16;
typedef bf16  bf16x4  __attribute__((ext_vector_type(4)));
typedef bf16  bf16x8  __attribute__((ext_vector_type(8)));
typedef bf16  bf16x16 __attribute__((ext_vector_type(16)));
typedef float f32x8   __attribute__((ext_vector_type(8)));

// ---------- CDNA5 async copy: global -> LDS, 16B per lane ----------
__device__ __forceinline__ void async_copy16(uint32_t lds_off, const void* gptr) {
    asm volatile("global_load_async_to_lds_b128 %0, %1, off"
                 :: "v"(lds_off), "v"(gptr) : "memory");
}
__device__ __forceinline__ void wait_async0() {
    asm volatile("s_wait_asynccnt 0x0" ::: "memory");
}
// Generic LDS pointer -> 32-bit LDS byte offset (low 32 bits of generic addr).
__device__ __forceinline__ uint32_t lds_off(const void* p) {
    return (uint32_t)(uintptr_t)p;
}

// ---------- WMMA helper: D = A(16x32 bf16) x B(32x16 bf16) + C(f32) ----------
__device__ __forceinline__ f32x8 wmma_bf16(bf16x16 a, bf16x16 b, f32x8 c) {
    return __builtin_amdgcn_wmma_f32_16x16x32_bf16(
        /*neg_a=*/false, a, /*neg_b=*/false, b,
        /*c_mod=*/(short)0, c, /*reuse_a=*/false, /*reuse_b=*/false);
}

// Load one 16x32 bf16 fragment (A or B operand; identical per-lane layouts).
// Lane L: row = L%16 (caller passes it), hh = L/16 selects the K interleave.
// Elements [0..7]  <- k = 8*hh + [0..8) ; [8..15] <- k = 16 + 8*hh + [0..8)
__device__ __forceinline__ bf16x16 load_frag(const bf16* base, int ld, int row, int hh) {
    const bf16* p = base + (size_t)row * (size_t)ld + hh * 8;
    bf16x8 lo = *(const bf16x8*)(p);
    bf16x8 hi = *(const bf16x8*)(p + 16);
    return __builtin_shufflevector(lo, hi, 0,1,2,3,4,5,6,7,8,9,10,11,12,13,14,15);
}

// ---------- f32 -> bf16 elementwise (activations) ----------
__global__ void cvt_f32_bf16_x4(const float4* __restrict__ in,
                                bf16* __restrict__ out, int n4) {
    int i = blockIdx.x * blockDim.x + threadIdx.x;
    if (i < n4) {
        float4 v = in[i];
        bf16x4 o;
        o[0] = (bf16)v.x; o[1] = (bf16)v.y; o[2] = (bf16)v.z; o[3] = (bf16)v.w;
        *(bf16x4*)(out + (size_t)i * 4) = o;
    }
}

// ---------- W (KxN f32) -> Wt (NxK bf16) tiled transpose ----------
__global__ __launch_bounds__(256)
void transpose_cvt(const float* __restrict__ W, bf16* __restrict__ Wt,
                   int K, int N) {
    __shared__ float tile[32][33];
    int n0 = blockIdx.x * 32, k0 = blockIdx.y * 32;
    int tx = threadIdx.x, ty = threadIdx.y;      // blockDim (32,8)
#pragma unroll
    for (int r = 0; r < 32; r += 8)
        tile[ty + r][tx] = W[(size_t)(k0 + ty + r) * N + (n0 + tx)];
    __syncthreads();
#pragma unroll
    for (int r = 0; r < 32; r += 8)
        Wt[(size_t)(n0 + ty + r) * K + (k0 + tx)] = (bf16)tile[tx][ty + r];
}

// ---------- GEMM: C(MxN) = A(MxK bf16) * Wt(NxK bf16)^T ----------
// WG: 256 thr = 8 waves, tile 128x128; wave tile 32x64 (2x4 WMMA tiles).
// A/W 128x64 k-slabs staged to LDS via async copies, double-buffered; two
// 32-wide WMMA k-steps per slab (one barrier per 64 k).
// MODE 0: f32 out[m*N+n] + bias[n]
// MODE 1: bf16 scatter to [B, HEADS, seq, DHEAD]  (Q with scale=0.125, K)
// MODE 2: bf16 scatter to [B, HEADS, DHEAD, seq]  (V transposed)
template <int MODE>
__global__ __launch_bounds__(256)
void gemm_wmma(const bf16* __restrict__ A, const bf16* __restrict__ Wt,
               void* __restrict__ out, const float* __restrict__ bias,
               int M, int N, int K, float scale) {
    __shared__ __align__(16) bf16 smA[2][128 * 64];
    __shared__ __align__(16) bf16 smW[2][128 * 64];
    const int t    = threadIdx.x;
    const int wid  = t >> 5;
    const int lane = t & 31;
    const int col  = lane & 15;
    const int hh   = lane >> 4;
    const int m_blk = blockIdx.x * 128, n_blk = blockIdx.y * 128;
    const int mw = (wid & 3) * 32;       // wave M offset within block tile
    const int nw = (wid >> 2) * 64;      // wave N offset within block tile

    // Stage 128x64 A and W k-slabs: 1024 16B chunks each, 4 per thread each.
    auto stage = [&](int buf, int k0) {
#pragma unroll
        for (int j = 0; j < 4; ++j) {
            const int c = t + 256 * j, row = c >> 3, cc = (c & 7) * 8;
            async_copy16(lds_off(&smA[buf][0]) + c * 16,
                         A + (size_t)(m_blk + row) * K + k0 + cc);
            async_copy16(lds_off(&smW[buf][0]) + c * 16,
                         Wt + (size_t)(n_blk + row) * K + k0 + cc);
        }
    };

    f32x8 acc[2][4];
#pragma unroll
    for (int i = 0; i < 2; ++i)
#pragma unroll
        for (int j = 0; j < 4; ++j)
            acc[i][j] = (f32x8){0.f,0.f,0.f,0.f,0.f,0.f,0.f,0.f};

    stage(0, 0);
    for (int k0 = 0; k0 < K; k0 += 64) {
        const int cur = (k0 >> 6) & 1;
        wait_async0();          // our async copies for 'cur' done
        __syncthreads();        // visible to all waves; old buffer fully consumed
        if (k0 + 64 < K) stage(cur ^ 1, k0 + 64);

#pragma unroll
        for (int ks = 0; ks < 64; ks += 32) {
            bf16x16 af[2], bfg[4];
#pragma unroll
            for (int i = 0; i < 2; ++i)
                af[i] = load_frag(&smA[cur][ks], 64, mw + i * 16 + col, hh);
#pragma unroll
            for (int j = 0; j < 4; ++j)
                bfg[j] = load_frag(&smW[cur][ks], 64, nw + j * 16 + col, hh);
#pragma unroll
            for (int i = 0; i < 2; ++i)
#pragma unroll
                for (int j = 0; j < 4; ++j)
                    acc[i][j] = wmma_bf16(af[i], bfg[j], acc[i][j]);
        }
    }

    // Epilogue. Accumulator: lane holds column n, rows m = base + r + 8*hh.
#pragma unroll
    for (int i = 0; i < 2; ++i) {
#pragma unroll
        for (int j = 0; j < 4; ++j) {
            const int n     = n_blk + nw + j * 16 + col;
            const int mbase = m_blk + mw + i * 16 + 8 * hh;
            if (MODE == 0) {
                float* o  = (float*)out;
                float  bv = bias[n];
#pragma unroll
                for (int r = 0; r < 8; ++r)
                    o[(size_t)(mbase + r) * N + n] = acc[i][j][r] + bv;
            } else if (MODE == 1) {
                bf16* o = (bf16*)out;
                const int hN = n >> 6, d = n & 63;
#pragma unroll
                for (int r = 0; r < 8; ++r) {
                    const int m = mbase + r;
                    const int b = m >> 11;          // / SQ (2048)
                    const int s = m & (SQ - 1);
                    o[(((size_t)(b * HEADS + hN) * SQ) + s) * DHEAD + d] =
                        (bf16)(acc[i][j][r] * scale);
                }
            } else { // MODE == 2: V transposed [B,H,D,Skv]
                bf16* o = (bf16*)out;
                const int hN = n >> 6, d = n & 63;
#pragma unroll
                for (int r = 0; r < 8; ++r) {
                    const int m = mbase + r;
                    const int b = m >> 11;
                    const int s = m & (SKV - 1);
                    o[(((size_t)(b * HEADS + hN) * DHEAD) + d) * SKV + s] =
                        (bf16)acc[i][j][r];
                }
            }
        }
    }
}

// ---------- Flash attention ----------
// Grid: (B*HEADS, SQ/(16*NW)), block 32*NW. Each wave owns 16 query rows; all
// 8 waves share (b,h), so 64-kv K/V slabs are staged once per WG into LDS
// (async copies, double-buffered). Computes S^T = K * Q^T so the score
// accumulator (lane = query column, kv rows in VGPRs) IS the B-fragment
// layout of P^T for O^T = V^T * P^T. One softmax/rescale per 64 kv.
#define NW 8
__global__ __launch_bounds__(32 * NW)
void flash_attn(const bf16* __restrict__ Q, const bf16* __restrict__ Kmat,
                const bf16* __restrict__ Vt, bf16* __restrict__ O) {
    __shared__ __align__(16) bf16 smK[2][64 * 64];   // 64 kv rows x 64 d
    __shared__ __align__(16) bf16 smV[2][64 * 64];   // 64 d rows x 64 kv
    const int t    = threadIdx.x;
    const int wid  = t >> 5;
    const int lane = t & 31;
    const int col  = lane & 15;
    const int hh   = lane >> 4;
    const int bh   = blockIdx.x;
    const int q0   = (blockIdx.y * NW + wid) * 16;

    const bf16* Qp = Q    + (size_t)bh * SQ * DHEAD;
    const bf16* Kp = Kmat + (size_t)bh * SKV * DHEAD;
    const bf16* Vp = Vt   + (size_t)bh * DHEAD * SKV;

    // K slab contiguous in global ([kv][d], 8KB); V slab = 64 rows of 128B.
    auto stage = [&](int buf, int kv0) {
#pragma unroll
        for (int j = 0; j < 2; ++j) {
            const int c = t + 256 * j;
            async_copy16(lds_off(&smK[buf][0]) + c * 16,
                         Kp + (size_t)kv0 * DHEAD + c * 8);
        }
#pragma unroll
        for (int j = 0; j < 2; ++j) {
            const int c = t + 256 * j, row = c >> 3, cc = (c & 7) * 8;
            async_copy16(lds_off(&smV[buf][0]) + c * 16,
                         Vp + (size_t)row * SKV + kv0 + cc);
        }
    };

    // Q B-fragments (lane = query column q0+col), kept resident.
    bf16x16 qf[2];
    qf[0] = load_frag(Qp + 0,  DHEAD, q0 + col, hh);
    qf[1] = load_frag(Qp + 32, DHEAD, q0 + col, hh);

    f32x8 o[4];
#pragma unroll
    for (int tt = 0; tt < 4; ++tt) o[tt] = (f32x8){0.f,0.f,0.f,0.f,0.f,0.f,0.f,0.f};
    float m_i = -INFINITY, l_i = 0.f;

    stage(0, 0);
    for (int kv0 = 0; kv0 < SKV; kv0 += 64) {
        const int cur = (kv0 >> 6) & 1;
        wait_async0();
        __syncthreads();
        if (kv0 + 64 < SKV) stage(cur ^ 1, kv0 + 64);

        // S^T tiles: sc[rb] covers kv rows [kv0+16*rb, kv0+16*rb+16).
        f32x8 sc[4];
#pragma unroll
        for (int rb = 0; rb < 4; ++rb) sc[rb] = (f32x8){0.f,0.f,0.f,0.f,0.f,0.f,0.f,0.f};
#pragma unroll
        for (int ds = 0; ds < 2; ++ds) {
            bf16x16 kf[4];
#pragma unroll
            for (int rb = 0; rb < 4; ++rb)
                kf[rb] = load_frag(&smK[cur][ds * 32], DHEAD, rb * 16 + col, hh);
#pragma unroll
            for (int rb = 0; rb < 4; ++rb)
                sc[rb] = wmma_bf16(kf[rb], qf[ds], sc[rb]);
        }
        // Lane holds, for query q = q0+col: kv = kv0 + rb*16 + r + 8*hh.
        // Pair (L, L^16) covers all 64 kv. Tree reductions for short dep chains.
        float t8[8];
#pragma unroll
        for (int r = 0; r < 8; ++r)
            t8[r] = fmaxf(fmaxf(sc[0][r], sc[1][r]), fmaxf(sc[2][r], sc[3][r]));
        float t4a = fmaxf(t8[0], t8[4]), t4b = fmaxf(t8[1], t8[5]);
        float t4c = fmaxf(t8[2], t8[6]), t4d = fmaxf(t8[3], t8[7]);
        float bm  = fmaxf(fmaxf(t4a, t4b), fmaxf(t4c, t4d));
        bm = fmaxf(bm, __shfl_xor(bm, 16));
        const float mn    = fmaxf(m_i, bm);
        const float alpha = __expf(m_i - mn);

        float e[4][8];
#pragma unroll
        for (int rb = 0; rb < 4; ++rb)
#pragma unroll
            for (int r = 0; r < 8; ++r)
                e[rb][r] = __expf(sc[rb][r] - mn);
        float p8[8];
#pragma unroll
        for (int r = 0; r < 8; ++r)
            p8[r] = (e[0][r] + e[1][r]) + (e[2][r] + e[3][r]);
        float p4a = p8[0] + p8[4], p4b = p8[1] + p8[5];
        float p4c = p8[2] + p8[6], p4d = p8[3] + p8[7];
        float psum = (p4a + p4b) + (p4c + p4d);
        psum += __shfl_xor(psum, 16);
        m_i = mn;
        l_i = l_i * alpha + psum;

        // P^T accumulator == B-fragment of P^T: direct bf16 repack, no shuffles.
        bf16x16 pf01, pf23;
#pragma unroll
        for (int r = 0; r < 8; ++r) {
            pf01[r]     = (bf16)e[0][r];
            pf01[8 + r] = (bf16)e[1][r];
            pf23[r]     = (bf16)e[2][r];
            pf23[8 + r] = (bf16)e[3][r];
        }
#pragma unroll
        for (int tt = 0; tt < 4; ++tt) o[tt] *= alpha;
#pragma unroll
        for (int tt = 0; tt < 4; ++tt) {
            bf16x16 vf = load_frag(&smV[cur][0], 64, tt * 16 + col, hh);
            o[tt] = wmma_bf16(vf, pf01, o[tt]);
        }
#pragma unroll
        for (int tt = 0; tt < 4; ++tt) {
            bf16x16 vf = load_frag(&smV[cur][32], 64, tt * 16 + col, hh);
            o[tt] = wmma_bf16(vf, pf23, o[tt]);
        }
    }

    // O^T accumulator: lane = query q0+col, rows d = tt*16 + r + 8*hh.
    const float inv = 1.f / l_i;
    const int b  = bh / HEADS;
    const int hN = bh % HEADS;
    const int s  = q0 + col;
    bf16* orow = O + ((size_t)b * SQ + s) * CDIM + hN * DHEAD;
#pragma unroll
    for (int tt = 0; tt < 4; ++tt) {
        bf16x8 pack;
#pragma unroll
        for (int r = 0; r < 8; ++r) pack[r] = (bf16)(o[tt][r] * inv);
        *(bf16x8*)(orow + tt * 16 + 8 * hh) = pack;
    }
}

// ---------- host ----------
extern "C" void kernel_launch(void* const* d_in, const int* in_sizes, int n_in,
                              void* d_out, int out_size, void* d_ws, size_t ws_size,
                              hipStream_t stream) {
    (void)in_sizes; (void)n_in; (void)out_size; (void)ws_size;
    const float* hs  = (const float*)d_in[0];   // (2,2048,1280)
    const float* ehs = (const float*)d_in[1];   // (2,2048,1024)
    const float* Wq  = (const float*)d_in[2];   // (1280,1280)
    const float* Wk  = (const float*)d_in[3];   // (1024,1280)
    const float* Wv  = (const float*)d_in[4];   // (1024,1280)
    const float* Wo  = (const float*)d_in[5];   // (1280,1280)
    const float* bo  = (const float*)d_in[6];   // (1280,)
    float*       out = (float*)d_out;

    char*  ws  = (char*)d_ws;
    size_t off = 0;
    auto alloc = [&](size_t bytes) -> void* {
        void* p = ws + off;
        off += (bytes + 255) & ~(size_t)255;
        return p;
    };
    bf16* hsb  = (bf16*)alloc((size_t)MROWS * CDIM  * 2);   // hs  bf16
    bf16* ehsb = (bf16*)alloc((size_t)MROWS * CCDIM * 2);   // ehs bf16
    bf16* Wqt  = (bf16*)alloc((size_t)CDIM  * CDIM  * 2);   // (1280x1280) N x K
    bf16* Wkt  = (bf16*)alloc((size_t)CDIM  * CCDIM * 2);   // (1280x1024)
    bf16* Wvt  = (bf16*)alloc((size_t)CDIM  * CCDIM * 2);
    bf16* Wot  = (bf16*)alloc((size_t)CDIM  * CDIM  * 2);
    bf16* Qb   = (bf16*)alloc((size_t)MROWS * CDIM  * 2);   // [B,H,Sq,D]
    bf16* Kb   = (bf16*)alloc((size_t)MROWS * CDIM  * 2);   // [B,H,Skv,D]
    bf16* Vtb  = (bf16*)alloc((size_t)MROWS * CDIM  * 2);   // [B,H,D,Skv]
    bf16* Ob   = (bf16*)alloc((size_t)MROWS * CDIM  * 2);   // attn out (B,Sq,C)

    // 1) activations -> bf16
    {
        int n4 = (MROWS * CDIM) / 4;
        cvt_f32_bf16_x4<<<(n4 + 255) / 256, 256, 0, stream>>>((const float4*)hs, hsb, n4);
        n4 = (MROWS * CCDIM) / 4;
        cvt_f32_bf16_x4<<<(n4 + 255) / 256, 256, 0, stream>>>((const float4*)ehs, ehsb, n4);
    }
    // 2) weights -> transposed bf16 (N x K)
    {
        dim3 blk(32, 8);
        transpose_cvt<<<dim3(CDIM / 32, CDIM  / 32), blk, 0, stream>>>(Wq, Wqt, CDIM,  CDIM);
        transpose_cvt<<<dim3(CDIM / 32, CCDIM / 32), blk, 0, stream>>>(Wk, Wkt, CCDIM, CDIM);
        transpose_cvt<<<dim3(CDIM / 32, CCDIM / 32), blk, 0, stream>>>(Wv, Wvt, CCDIM, CDIM);
        transpose_cvt<<<dim3(CDIM / 32, CDIM  / 32), blk, 0, stream>>>(Wo, Wot, CDIM,  CDIM);
    }
    // 3) projections (Q gets the 1/sqrt(64) scale folded in)
    {
        dim3 grid(MROWS / 128, CDIM / 128);
        gemm_wmma<1><<<grid, 256, 0, stream>>>(hsb,  Wqt, Qb,  nullptr, MROWS, CDIM, CDIM,  0.125f);
        gemm_wmma<1><<<grid, 256, 0, stream>>>(ehsb, Wkt, Kb,  nullptr, MROWS, CDIM, CCDIM, 1.0f);
        gemm_wmma<2><<<grid, 256, 0, stream>>>(ehsb, Wvt, Vtb, nullptr, MROWS, CDIM, CCDIM, 1.0f);
    }
    // 4) flash attention (bf16 in/out, f32 softmax & accumulation)
    flash_attn<<<dim3(BATCH * HEADS, SQ / (16 * NW)), 32 * NW, 0, stream>>>(Qb, Kb, Vtb, Ob);
    // 5) output projection + bias -> f32
    {
        dim3 grid(MROWS / 128, CDIM / 128);
        gemm_wmma<0><<<grid, 256, 0, stream>>>(Ob, Wot, out, bo, MROWS, CDIM, CDIM, 1.0f);
    }
}
